// SpatioTemporalTransformer_70145405878951
// MI455X (gfx1250) — matrix operand order repaired and checked by reading, not compiled, not run on hardware
//
#include <hip/hip_runtime.h>
#include <hip/hip_bf16.h>
#include <stdint.h>

// ---- problem dims ----
#define B_   8
#define T_   128
#define N_   24
#define D_   128
#define M_   9
#define H_   8
#define L_   4
#define FF_  1024
#define F_   16        // D/H
#define ND_  3072      // N*D
#define NM_  216       // N*M
#define S_   1024      // T*B tokens, token id s = t*B + b

typedef __attribute__((ext_vector_type(16))) _Float16 v16h;
typedef __attribute__((ext_vector_type(8)))  _Float16 v8h;
typedef __attribute__((ext_vector_type(8)))  float    v8f;

union AFrag { v16h v; v8h h[2]; _Float16 e[16]; };

// ---- CDNA5 async global->LDS copy (ASYNCcnt path) ----
// VDST operand = VGPR holding LDS byte offset; flat shared pointers are
// {SHARED_BASE, lds_offset[31:0]}, so truncating the flat address gives the offset.
__device__ __forceinline__ void async_g2l_b32(void* lds_ptr, const void* gptr)
{
    unsigned lds_off = (unsigned)(uintptr_t)lds_ptr;
    asm volatile("global_load_async_to_lds_b32 %0, %1, off"
                 :: "v"(lds_off), "v"(gptr) : "memory");
}
__device__ __forceinline__ void wait_async0()
{
    asm volatile("s_wait_asynccnt 0x0" ::: "memory");
}

// =====================================================================
// Generic batched WMMA GEMM:  C[z][M x Nout] = A_z[M x K] * W_z[Nout x K]^T + bias_z
//   A: f16, row stride a_rs, per-batch column offset (z % a_mod) * a_cs
//   W: f16 row-major (NoutPad x K), batch offset (z / w_div) * w_bs
//   bias: f32, offset (z / w_div) * bias_s
//   C: f32, row stride c_rs, batch offset z * c_bs
// One wave computes one 16x16 tile; 4 waves per block tile the M dim.
// =====================================================================
__global__ __launch_bounds__(128) void k_gemm(
    const _Float16* __restrict__ A, int a_rs, int a_mod, int a_cs,
    const _Float16* __restrict__ W, int w_div, long w_bs,
    const float* __restrict__ bias, int bias_s,
    float* __restrict__ C, long c_rs, long c_bs,
    int M, int Nout, int K)
{
    const int z = blockIdx.z;
    const _Float16* Az = A + (long)(z % a_mod) * a_cs;
    const _Float16* Wz = W + (long)(z / w_div) * w_bs;
    const float*    bz = bias + (long)(z / w_div) * bias_s;
    float*          Cz = C + (long)z * c_bs;

    const int lane = threadIdx.x;                       // 0..31
    const int row0 = (blockIdx.x * 4 + threadIdx.y) * 16;
    const int col0 = blockIdx.y * 16;
    if (row0 >= M) return;

    const int mr = lane & 15;
    const int hi = lane >> 4;

    const _Float16* arow = Az + (long)(row0 + mr) * a_rs;   // A row (this lane's M row)
    const _Float16* wrow = Wz + (long)(col0 + mr) * K;      // weight row = output column
    const int akoff = hi * 16;   // A frag: lanes 0-15: K 0..15 ; lanes 16-31: K 16..31
    const int bkoff = hi * 16;   // B frag: lanes 0-15: K 0..15 ; lanes 16-31: K 16..31

    v8f acc = {};
    for (int k0 = 0; k0 < K; k0 += 32) {
        AFrag af, bf;
        af.h[0] = *(const v8h*)(arow + k0 + akoff);
        af.h[1] = *(const v8h*)(arow + k0 + 8 + akoff);
        bf.v    = *(const v16h*)(wrow + k0 + bkoff);
        __builtin_prefetch(arow + k0 + 128, 0, 1);
        acc = __builtin_amdgcn_wmma_f32_16x16x32_f16(
            false, af.v, false, bf.v, (short)0, acc, false, false);
    }

    const int cc = col0 + mr;                 // C: lane&15 = column
    if (cc < Nout) {
        const float bv = bz[cc];
        const int rbase = row0 + hi * 8;      // VGPR r = row r (+8 for hi lanes)
        for (int r = 0; r < 8; ++r)
            Cz[(long)(rbase + r) * c_rs + cc] = acc[r] + bv;
    }
}

// ---- f32 -> f16 elementwise convert (also used for weight copies) ----
__global__ void k_cvt16(const float* __restrict__ in, _Float16* __restrict__ out, long n)
{
    long i = (long)blockIdx.x * 256 + threadIdx.x;
    if (i < n) out[i] = (_Float16)in[i];
}

// ---- transpose last two dims + convert: in[b][r][c] -> out[b][cp][r], zero pad cp>=C ----
__global__ void k_tcvt(const float* __restrict__ in, _Float16* __restrict__ out,
                       int Bn, int R, int C, int Cpad)
{
    long total = (long)Bn * Cpad * R;
    long i = (long)blockIdx.x * 256 + threadIdx.x;
    if (i >= total) return;
    int  r  = (int)(i % R);
    long q  = i / R;
    int  c  = (int)(q % Cpad);
    int  b  = (int)(q / Cpad);
    out[i] = (c < C) ? (_Float16)in[((long)b * R + r) * C + c] : (_Float16)0.0f;
}

// ---- embedding + positional encoding -> h (T,B,ND) in f32 and f16 ----
__global__ __launch_bounds__(128) void k_embed(
    const float* __restrict__ x, const float* __restrict__ eW,
    const float* __restrict__ eb, float* __restrict__ h32, _Float16* __restrict__ h16)
{
    const int blk = blockIdx.x;            // (b*T + t)*N + n
    const int n  = blk % N_;
    const int bt = blk / N_;
    const int t  = bt % T_;
    const int b  = bt / T_;
    __shared__ float xm[M_];
    if (threadIdx.x < M_) xm[threadIdx.x] = x[(long)blk * M_ + threadIdx.x];
    __syncthreads();
    const int d = threadIdx.x;             // 0..127
    float acc = eb[n * D_ + d];
    for (int m = 0; m < M_; ++m) acc += xm[m] * eW[(n * M_ + m) * D_ + d];
    const int i = n * D_ + d;
    const float dv  = expf(-(float)(i & ~1) * (logf(10000.0f) / (float)ND_));
    const float arg = (float)b * dv;
    acc += (i & 1) ? cosf(arg) : sinf(arg);
    const long o = (long)(t * B_ + b) * ND_ + i;
    h32[o] = acc;
    h16[o] = (_Float16)acc;
}

// ---- spatial attention: per (token s, head h): softmax over N=24 nodes ----
__global__ __launch_bounds__(128) void k_sp_attn(
    const float* __restrict__ qb, const float* __restrict__ kb,
    const float* __restrict__ vb, float* __restrict__ spr)
{
    const int s = blockIdx.x, h = blockIdx.y;
    const long base = ((long)s * H_ + h) * N_ * F_;
    __shared__ float q[N_ * F_], k[N_ * F_], v[N_ * F_], a[N_ * N_];
    for (int i = threadIdx.x; i < N_ * F_; i += 128) {
        async_g2l_b32(&q[i], qb + base + i);
        async_g2l_b32(&k[i], kb + base + i);
        async_g2l_b32(&v[i], vb + base + i);
    }
    wait_async0();
    __syncthreads();
    for (int idx = threadIdx.x; idx < N_ * N_; idx += 128) {
        int n = idx / N_, m = idx % N_;
        float d = 0.f;
        for (int f = 0; f < F_; ++f) d += q[n * F_ + f] * k[m * F_ + f];
        a[idx] = d * 0.25f;                       // 1/sqrt(F=16)
    }
    __syncthreads();
    if (threadIdx.x < N_) {
        int n = threadIdx.x;
        float mx = -1e30f;
        for (int m = 0; m < N_; ++m) mx = fmaxf(mx, a[n * N_ + m]);
        float sm = 0.f;
        for (int m = 0; m < N_; ++m) { float e = expf(a[n * N_ + m] - mx); a[n * N_ + m] = e; sm += e; }
        float inv = 1.f / sm;
        for (int m = 0; m < N_; ++m) a[n * N_ + m] *= inv;
    }
    __syncthreads();
    for (int idx = threadIdx.x; idx < N_ * F_; idx += 128) {
        int n = idx / F_, f = idx % F_;
        float o = 0.f;
        for (int m = 0; m < N_; ++m) o += a[n * N_ + m] * v[m * F_ + f];
        spr[(long)s * ND_ + n * D_ + h * F_ + f] = o;   // (t,b,n,h,f) flat
    }
}

// ---- temporal attention per (b,n,h): T=128, HD=16, additive +1 mask for s<t ----
__global__ __launch_bounds__(128) void k_tmp_attn(
    const float* __restrict__ qkvt, float* __restrict__ ot)
{
    const int bn = blockIdx.x;
    const int b = bn / N_, n = bn % N_, h = blockIdx.y;
    __shared__ float kk[T_ * F_], vv[T_ * F_];
    for (int i = threadIdx.x; i < T_ * F_; i += 128) {
        int t2 = i >> 4, f = i & 15;
        long rb = ((long)(t2 * B_ + b) * N_ + n) * (3 * D_) + h * F_ + f;
        async_g2l_b32(&kk[i], qkvt + rb + D_);
        async_g2l_b32(&vv[i], qkvt + rb + 2 * D_);
    }
    wait_async0();
    __syncthreads();
    const int t = threadIdx.x;
    float q[F_];
    const long qoff = ((long)(t * B_ + b) * N_ + n) * (3 * D_) + h * F_;
    for (int f = 0; f < F_; ++f) q[f] = qkvt[qoff + f];
    float m = -1e30f, lsum = 0.f, acc[F_];
    for (int f = 0; f < F_; ++f) acc[f] = 0.f;
    for (int s2 = 0; s2 < T_; ++s2) {
        float d = 0.f;
        for (int f = 0; f < F_; ++f) d += q[f] * kk[s2 * F_ + f];
        d = d * 0.25f + ((s2 < t) ? 1.0f : 0.0f);   // 1/sqrt(HD=16) + tril(-1) mask
        float nm = fmaxf(m, d);
        float sc = expf(m - nm);
        float e  = expf(d - nm);
        lsum = lsum * sc + e;
        for (int f = 0; f < F_; ++f) acc[f] = acc[f] * sc + e * vv[s2 * F_ + f];
        m = nm;
    }
    const float inv = 1.f / lsum;
    const long ob = (long)(t * B_ + b) * ND_ + n * D_ + h * F_;
    for (int f = 0; f < F_; ++f) ot[ob + f] = acc[f] * inv;
}

// ---- y = LayerNorm(xa + xb)*g + b  (optionally also f16 copy); one block per token ----
__global__ __launch_bounds__(256) void k_add_ln(
    const float* __restrict__ xa, const float* __restrict__ xb,
    const float* __restrict__ g, const float* __restrict__ bt,
    float* __restrict__ y, _Float16* __restrict__ y16)
{
    __shared__ float red[256];
    const long base = (long)blockIdx.x * ND_;
    float s = 0.f;
    for (int i = threadIdx.x; i < ND_; i += 256) s += xa[base + i] + xb[base + i];
    red[threadIdx.x] = s; __syncthreads();
    for (int st = 128; st > 0; st >>= 1) {
        if (threadIdx.x < st) red[threadIdx.x] += red[threadIdx.x + st];
        __syncthreads();
    }
    const float mean = red[0] * (1.0f / ND_);
    __syncthreads();
    float v = 0.f;
    for (int i = threadIdx.x; i < ND_; i += 256) {
        float d = xa[base + i] + xb[base + i] - mean; v += d * d;
    }
    red[threadIdx.x] = v; __syncthreads();
    for (int st = 128; st > 0; st >>= 1) {
        if (threadIdx.x < st) red[threadIdx.x] += red[threadIdx.x + st];
        __syncthreads();
    }
    const float rstd = rsqrtf(red[0] * (1.0f / ND_) + 1e-5f);
    for (int i = threadIdx.x; i < ND_; i += 256) {
        float val = (xa[base + i] + xb[base + i] - mean) * rstd * g[i] + bt[i];
        y[base + i] = val;
        if (y16) y16[base + i] = (_Float16)val;
    }
}

// ---- o32 = a + b (also f16 copy) ----
__global__ void k_add2(const float* __restrict__ a, const float* __restrict__ b,
                       float* __restrict__ o, _Float16* __restrict__ o16, long n)
{
    long i = (long)blockIdx.x * 256 + threadIdx.x;
    if (i < n) { float v = a[i] + b[i]; o[i] = v; o16[i] = (_Float16)v; }
}

// ---- out[b][t][j] = fin[(t*B+b)][j] + inputs[b][t][j] ----
__global__ void k_final(const float* __restrict__ fin, const float* __restrict__ inp,
                        float* __restrict__ out)
{
    const int bt = blockIdx.x;   // b*T + t
    const int j  = threadIdx.x;
    if (j >= NM_) return;
    const int b = bt / T_, t = bt % T_;
    out[(long)bt * NM_ + j] = fin[(long)(t * B_ + b) * NM_ + j] + inp[(long)bt * NM_ + j];
}

// =====================================================================
// Host side
// =====================================================================
extern "C" void kernel_launch(void* const* d_in, const int* in_sizes, int n_in,
                              void* d_out, int out_size, void* d_ws, size_t ws_size,
                              hipStream_t stream)
{
    (void)in_sizes; (void)n_in; (void)out_size; (void)ws_size;

    const float* x_in     = (const float*)d_in[0];
    const float* emb_W    = (const float*)d_in[1];
    const float* emb_b    = (const float*)d_in[2];
    const float* sp_Wq    = (const float*)d_in[3];
    const float* sp_bq    = (const float*)d_in[4];
    const float* sp_Wk    = (const float*)d_in[5];
    const float* sp_bk    = (const float*)d_in[6];
    const float* sp_Wv    = (const float*)d_in[7];
    const float* sp_bv    = (const float*)d_in[8];
    const float* tmp_Win  = (const float*)d_in[9];
    const float* tmp_bin  = (const float*)d_in[10];
    const float* tmp_Wout = (const float*)d_in[11];
    const float* tmp_bout = (const float*)d_in[12];
    const float* ff_W1    = (const float*)d_in[13];
    const float* ff_b1    = (const float*)d_in[14];
    const float* ff_W2    = (const float*)d_in[15];
    const float* ff_b2    = (const float*)d_in[16];
    const float* ln_g     = (const float*)d_in[17];
    const float* ln_b     = (const float*)d_in[18];
    const float* fin_W    = (const float*)d_in[19];
    const float* fin_b    = (const float*)d_in[20];
    float* out = (float*)d_out;

    // ---- workspace layout (all offsets 256B aligned) ----
    char* ws = (char*)d_ws;
    const size_t OFF_H32  = 0;                           // S*ND f32      12582912
    const size_t OFF_H16  = OFF_H32  + 12582912;         // S*ND f16       6291456
    const size_t OFF_SP32 = OFF_H16  + 6291456;          // sp / a2 f32   12582912
    const size_t OFF_TP32 = OFF_SP32 + 12582912;         // tp / ff2 f32  12582912
    const size_t OFF_X16  = OFF_TP32 + 12582912;         // ot16 then a2_16 6291456
    const size_t OFF_OT32 = OFF_X16  + 6291456;          // ot f32        12582912
    const size_t OFF_U    = OFF_OT32 + 12582912;         // union region  37748736
    const size_t OFF_W1T  = OFF_U    + 37748736;         // L*FF*ND f16   25165824
    const size_t OFF_W2T  = OFF_W1T  + 25165824;         // L*ND*FF f16   25165824
    const size_t OFF_WIN  = OFF_W2T  + 25165824;         // L*N*3D*D f16   9437184
    const size_t OFF_WOUT = OFF_WIN  + 9437184;          // L*N*D*D f16    3145728
    const size_t OFF_SPQ  = OFF_WOUT + 3145728;          // L*H*N*F*D f16  3145728
    const size_t OFF_SPK  = OFF_SPQ  + 3145728;          // L*H*F*D f16     131072
    const size_t OFF_SPV  = OFF_SPK  + 131072;           //                 131072
    const size_t OFF_FINT = OFF_SPV  + 131072;           // 224*ND f16     1376256

    float*    h32   = (float*)(ws + OFF_H32);
    _Float16* h16   = (_Float16*)(ws + OFF_H16);
    float*    sp32  = (float*)(ws + OFF_SP32);
    float*    tp32  = (float*)(ws + OFF_TP32);
    _Float16* x16   = (_Float16*)(ws + OFF_X16);     // ot16, then a2_16
    float*    ot32  = (float*)(ws + OFF_OT32);
    float*    qbuf  = (float*)(ws + OFF_U);
    float*    kbuf  = (float*)(ws + OFF_U + 12582912);
    float*    vbuf  = (float*)(ws + OFF_U + 25165824);
    float*    qkvt  = (float*)(ws + OFF_U);
    float*    ff1_32 = (float*)(ws + OFF_U);
    _Float16* ff1_16 = (_Float16*)(ws + OFF_U + 4194304);
    float*    fin32 = (float*)(ws + OFF_U);
    _Float16* W1T  = (_Float16*)(ws + OFF_W1T);
    _Float16* W2T  = (_Float16*)(ws + OFF_W2T);
    _Float16* WIN  = (_Float16*)(ws + OFF_WIN);
    _Float16* WOUT = (_Float16*)(ws + OFF_WOUT);
    _Float16* SPQ  = (_Float16*)(ws + OFF_SPQ);
    _Float16* SPK  = (_Float16*)(ws + OFF_SPK);
    _Float16* SPV  = (_Float16*)(ws + OFF_SPV);
    _Float16* FINT = (_Float16*)(ws + OFF_FINT);

    const dim3 gblk(32, 4, 1);
    auto cdiv = [](long a, long b) { return (int)((a + b - 1) / b); };

    // ---- weight convert / transpose to f16 ----
    { long n = (long)L_ * N_ * 3 * D_ * D_; k_cvt16<<<cdiv(n,256),256,0,stream>>>(tmp_Win,  WIN,  n); }
    { long n = (long)L_ * N_ * D_ * D_;     k_cvt16<<<cdiv(n,256),256,0,stream>>>(tmp_Wout, WOUT, n); }
    k_tcvt<<<cdiv((long)L_*FF_*ND_,256),256,0,stream>>>(ff_W1, W1T, L_, ND_, FF_, FF_);
    k_tcvt<<<cdiv((long)L_*ND_*FF_,256),256,0,stream>>>(ff_W2, W2T, L_, FF_, ND_, ND_);
    k_tcvt<<<cdiv((long)L_*H_*N_*F_*D_,256),256,0,stream>>>(sp_Wq, SPQ, L_*H_*N_, D_, F_, F_);
    k_tcvt<<<cdiv((long)L_*H_*F_*D_,256),256,0,stream>>>(sp_Wk, SPK, L_*H_, D_, F_, F_);
    k_tcvt<<<cdiv((long)L_*H_*F_*D_,256),256,0,stream>>>(sp_Wv, SPV, L_*H_, D_, F_, F_);
    k_tcvt<<<cdiv((long)224*ND_,256),256,0,stream>>>(fin_W, FINT, 1, ND_, NM_, 224);

    // ---- embedding + positional encoding ----
    k_embed<<<B_*T_*N_, 128, 0, stream>>>(x_in, emb_W, emb_b, h32, h16);

    for (int l = 0; l < L_; ++l) {
        const _Float16* spq_l  = SPQ  + (long)l * H_ * N_ * F_ * D_;
        const _Float16* spk_l  = SPK  + (long)l * H_ * F_ * D_;
        const _Float16* spv_l  = SPV  + (long)l * H_ * F_ * D_;
        const _Float16* win_l  = WIN  + (long)l * N_ * 3 * D_ * D_;
        const _Float16* wout_l = WOUT + (long)l * N_ * D_ * D_;
        const _Float16* w1_l   = W1T  + (long)l * FF_ * ND_;
        const _Float16* w2_l   = W2T  + (long)l * ND_ * FF_;
        const float* g_l  = ln_g + (long)l * ND_;
        const float* b_l  = ln_b + (long)l * ND_;

        // ---- spatial branch: q/k/v projections (batched over h,n), WMMA ----
        k_gemm<<<dim3(S_/64, 1, H_*N_), gblk, 0, stream>>>(
            h16, ND_, N_, D_, spq_l, 1, (long)F_*D_,
            sp_bq + (long)l*H_*N_*F_, F_, qbuf, (long)H_*N_*F_, (long)F_, S_, F_, D_);
        k_gemm<<<dim3(S_/64, 1, H_*N_), gblk, 0, stream>>>(
            h16, ND_, N_, D_, spk_l, N_, (long)F_*D_,
            sp_bk + (long)l*H_*F_, F_, kbuf, (long)H_*N_*F_, (long)F_, S_, F_, D_);
        k_gemm<<<dim3(S_/64, 1, H_*N_), gblk, 0, stream>>>(
            h16, ND_, N_, D_, spv_l, N_, (long)F_*D_,
            sp_bv + (long)l*H_*F_, F_, vbuf, (long)H_*N_*F_, (long)F_, S_, F_, D_);
        k_sp_attn<<<dim3(S_, H_), 128, 0, stream>>>(qbuf, kbuf, vbuf, sp32);
        k_add_ln<<<S_, 256, 0, stream>>>(sp32, h32, g_l, b_l, sp32, (_Float16*)nullptr);

        // ---- temporal branch: fused qkv projection (batched over n), WMMA ----
        k_gemm<<<dim3(S_/64, (3*D_)/16, N_), gblk, 0, stream>>>(
            h16, ND_, N_, D_, win_l, 1, (long)3*D_*D_,
            tmp_bin + (long)l*N_*3*D_, 3*D_, qkvt, (long)N_*3*D_, (long)3*D_, S_, 3*D_, D_);
        k_tmp_attn<<<dim3(B_*N_, H_), 128, 0, stream>>>(qkvt, ot32);
        { long n = (long)S_*ND_; k_cvt16<<<cdiv(n,256),256,0,stream>>>(ot32, x16, n); }
        k_gemm<<<dim3(S_/64, D_/16, N_), gblk, 0, stream>>>(
            x16, ND_, N_, D_, wout_l, 1, (long)D_*D_,
            tmp_bout + (long)l*N_*D_, D_, tp32, (long)ND_, (long)D_, S_, D_, D_);
        k_add_ln<<<S_, 256, 0, stream>>>(tp32, h32, g_l, b_l, tp32, (_Float16*)nullptr);

        // ---- a2 = sp + tp (f32 residual + f16 for FF input) ----
        { long n = (long)S_*ND_; k_add2<<<cdiv(n,256),256,0,stream>>>(sp32, tp32, sp32, x16, n); }

        // ---- FF: (a2 @ W1 + b1) @ W2 + b2, both WMMA ----
        k_gemm<<<dim3(S_/64, FF_/16, 1), gblk, 0, stream>>>(
            x16, ND_, 1, 0, w1_l, 1, 0L,
            ff_b1 + (long)l*FF_, 0, ff1_32, (long)FF_, 0L, S_, FF_, ND_);
        { long n = (long)S_*FF_; k_cvt16<<<cdiv(n,256),256,0,stream>>>(ff1_32, ff1_16, n); }
        k_gemm<<<dim3(S_/64, ND_/16, 1), gblk, 0, stream>>>(
            ff1_16, FF_, 1, 0, w2_l, 1, 0L,
            ff_b2 + (long)l*ND_, 0, tp32, (long)ND_, 0L, S_, ND_, FF_);

        // ---- h = LN(ff_out + a2), keep f32 + f16 ----
        k_add_ln<<<S_, 256, 0, stream>>>(tp32, sp32, g_l, b_l, h32, h16);
    }

    // ---- final projection (Nout=216 padded to 224) + residual input add ----
    k_gemm<<<dim3(S_/64, 224/16, 1), gblk, 0, stream>>>(
        h16, ND_, 1, 0, FINT, 1, 0L,
        fin_b, 0, fin32, (long)NM_, 0L, S_, NM_, ND_);
    k_final<<<B_*T_, 224, 0, stream>>>(fin32, x_in, out);
}
